// CausalGatedD2Attention_86045374808073
// MI455X (gfx1250) — compile-verified, matrix operand order checked by reading, not run on hardware
//
#include <hip/hip_runtime.h>
#include <hip/hip_bf16.h>

#define D_MODEL 1024
#define SEQ_T   2048
#define BATCH   4
#define BT      (BATCH * SEQ_T)

typedef __attribute__((ext_vector_type(16))) __bf16 v16bf;
typedef __attribute__((ext_vector_type(8)))  float  v8f;

union Frag {
    v16bf v;
    unsigned short u[16];
    uint4 q[2];
};

__device__ __forceinline__ unsigned short f2bf(float f) {
    unsigned int u = __float_as_uint(f);
    u += 0x7fffu + ((u >> 16) & 1u);   // round-to-nearest-even
    return (unsigned short)(u >> 16);
}

__device__ __forceinline__ v8f wmma_bf16(const Frag& a, const Frag& b, v8f c) {
    // D = A(16x32 bf16) * B(32x16 bf16) + C(16x16 f32)
    return __builtin_amdgcn_wmma_f32_16x16x32_bf16(
        /*neg_a=*/false, a.v, /*neg_b=*/false, b.v,
        /*c_mod=*/(short)0, c, /*reuse_a=*/false, /*reuse_b=*/false);
}

// ---------------------------------------------------------------------------
// Kernel 1: fp32 -> bf16 weight conversion
// ---------------------------------------------------------------------------
__global__ __launch_bounds__(256)
void cvt_bf16_kernel(const float* __restrict__ src, unsigned short* __restrict__ dst, int n) {
    int i = blockIdx.x * 256 + threadIdx.x;
    if (i < n) dst[i] = f2bf(src[i]);
}

// ---------------------------------------------------------------------------
// Kernel 2: LayerNorm -> bf16  (one block per row of 1024)
// ---------------------------------------------------------------------------
__global__ __launch_bounds__(256)
void layernorm_bf16_kernel(const float* __restrict__ x, const float* __restrict__ g,
                           const float* __restrict__ bta, unsigned short* __restrict__ xnb) {
    __shared__ float red[256];
    __shared__ float red2[256];
    const int row = blockIdx.x;
    const int tid = threadIdx.x;
    const float* xr = x + (size_t)row * D_MODEL;

    float4 vv = *(const float4*)&xr[tid * 4];
    float v0 = vv.x, v1 = vv.y, v2 = vv.z, v3 = vv.w;
    red[tid]  = v0 + v1 + v2 + v3;
    red2[tid] = v0*v0 + v1*v1 + v2*v2 + v3*v3;
    __syncthreads();
    for (int off = 128; off > 0; off >>= 1) {
        if (tid < off) { red[tid] += red[tid + off]; red2[tid] += red2[tid + off]; }
        __syncthreads();
    }
    const float mu   = red[0] * (1.0f / D_MODEL);
    const float var  = red2[0] * (1.0f / D_MODEL) - mu * mu;
    const float rstd = rsqrtf(var + 1e-5f);

    float vr[4] = {v0, v1, v2, v3};
    #pragma unroll
    for (int j = 0; j < 4; ++j) {
        int c = tid * 4 + j;
        xnb[(size_t)row * D_MODEL + c] = f2bf((vr[j] - mu) * rstd * g[c] + bta[c]);
    }
}

// ---------------------------------------------------------------------------
// Kernel 3: fused QKV + gate projection GEMM (bf16 WMMA, f32 accum)
// grid = (BT/64, D/64, 3)  mode z: 0=q(elu+1) 1=k(gate+elu+1, two GEMMs) 2=v
// ---------------------------------------------------------------------------
__global__ __launch_bounds__(256)
void qkvgate_gemm_kernel(const unsigned short* __restrict__ xnb,
                         const unsigned short* __restrict__ wqkv,   // [1024][3072] bf16
                         const unsigned short* __restrict__ wgate,  // [1024][1024] bf16
                         const float* __restrict__ qkv_b,
                         const float* __restrict__ gate_b,
                         unsigned short* __restrict__ qb,
                         unsigned short* __restrict__ kb,
                         unsigned short* __restrict__ vb) {
    __shared__ __align__(16) unsigned short As[64 * 32];
    __shared__ __align__(16) unsigned short Bs[64 * 32];   // transposed: [n][k]
    __shared__ __align__(16) unsigned short Gs[64 * 32];   // transposed gate weights

    const int rowBase = blockIdx.x * 64;
    const int colBase = blockIdx.y * 64;
    const int mode    = blockIdx.z;
    const int tid  = threadIdx.x;
    const int w    = tid >> 5, lane = tid & 31;
    const int rW   = w & 3,  cW   = w >> 2;
    const int h    = lane >> 4, ln16 = lane & 15;
    const int colW = (mode == 0) ? colBase
                   : (mode == 1) ? (D_MODEL + colBase) : (2 * D_MODEL + colBase);

    v8f acc0 = {}, acc1 = {}, accG0 = {}, accG1 = {};

    for (int kk = 0; kk < D_MODEL / 32; ++kk) {
        { // stage A: 64x32, 8 contiguous bf16 per thread
            int lin = tid * 8;
            int r = lin >> 5, c = lin & 31;
            *(uint4*)&As[lin] =
                *(const uint4*)&xnb[(size_t)(rowBase + r) * D_MODEL + kk * 32 + c];
        }
        #pragma unroll
        for (int i = 0; i < 8; ++i) { // stage B transposed (coalesced over n)
            int lin = tid + 256 * i;
            int kd = lin >> 6, n = lin & 63;
            Bs[n * 32 + kd] = wqkv[(size_t)(kk * 32 + kd) * (3 * D_MODEL) + colW + n];
            if (mode == 1)
                Gs[n * 32 + kd] = wgate[(size_t)(kk * 32 + kd) * D_MODEL + colBase + n];
        }
        __syncthreads();

        Frag a, b0, b1;
        const int mrow = 16 * rW + ln16;
        a.q[0] = *(const uint4*)&As[mrow * 32 + 8 * h];
        a.q[1] = *(const uint4*)&As[mrow * 32 + 16 + 8 * h];
        const int n0 = 32 * cW + ln16;
        const int n1 = n0 + 16;
        b0.q[0] = *(const uint4*)&Bs[n0 * 32 + 8 * h];
        b0.q[1] = *(const uint4*)&Bs[n0 * 32 + 16 + 8 * h];
        b1.q[0] = *(const uint4*)&Bs[n1 * 32 + 8 * h];
        b1.q[1] = *(const uint4*)&Bs[n1 * 32 + 16 + 8 * h];
        acc0 = wmma_bf16(a, b0, acc0);
        acc1 = wmma_bf16(a, b1, acc1);
        if (mode == 1) {
            Frag g0, g1;
            g0.q[0] = *(const uint4*)&Gs[n0 * 32 + 8 * h];
            g0.q[1] = *(const uint4*)&Gs[n0 * 32 + 16 + 8 * h];
            g1.q[0] = *(const uint4*)&Gs[n1 * 32 + 8 * h];
            g1.q[1] = *(const uint4*)&Gs[n1 * 32 + 16 + 8 * h];
            accG0 = wmma_bf16(a, g0, accG0);
            accG1 = wmma_bf16(a, g1, accG1);
        }
        __syncthreads();
    }

    // epilogue: bias + activation, write bf16
    #pragma unroll
    for (int t = 0; t < 2; ++t) {
        v8f acc  = t ? acc1 : acc0;
        v8f accG = t ? accG1 : accG0;
        const int ncol = colBase + 32 * cW + 16 * t + ln16;
        const float bias = (mode == 0) ? qkv_b[ncol]
                         : (mode == 1) ? qkv_b[D_MODEL + ncol]
                                       : qkv_b[2 * D_MODEL + ncol];
        const float gbias = (mode == 1) ? gate_b[ncol] : 0.0f;
        #pragma unroll
        for (int r = 0; r < 8; ++r) {
            const int mg = rowBase + 16 * rW + r + 8 * h;
            const size_t idx = (size_t)mg * D_MODEL + ncol;
            const float val = acc[r] + bias;
            if (mode == 0) {
                float qv = val > 0.0f ? val + 1.0f : __expf(val); // elu(x)+1
                qb[idx] = f2bf(qv);
            } else if (mode == 1) {
                float g  = 1.0f / (1.0f + __expf(-(accG[r] + gbias)));
                float kg = val * g;
                float kv = kg > 0.0f ? kg + 1.0f : __expf(kg);
                kb[idx] = f2bf(kv);
            } else {
                vb[idx] = f2bf(val);
            }
        }
    }
}

// ---------------------------------------------------------------------------
// Kernel 4: causal linear attention, flash-style over key chunks.
// grid = (T/64, D/64, B). Block owns out rows [t0,t0+64) x cols [d0,d0+64).
// den (rowsum of masked q.kT == q . k_cumsum) accumulated via LDS ds_add_f32.
// ---------------------------------------------------------------------------
__global__ __launch_bounds__(256)
void linattn_kernel(const unsigned short* __restrict__ qb,
                    const unsigned short* __restrict__ kb,
                    const unsigned short* __restrict__ vb,
                    float* __restrict__ out) {
    __shared__ __align__(16) unsigned short Qs[64 * 32];
    __shared__ __align__(16) unsigned short Ks[64 * 32];   // k rows are already [s][d] = Bt layout
    __shared__ __align__(16) unsigned short Aw[64 * 64];   // masked attn tile, bf16
    __shared__ __align__(16) unsigned short Vs[64 * 32];   // v staged transposed [n][s]
    __shared__ float den[64];

    const int tTile = blockIdx.x;
    const int t0 = tTile * 64;
    const int d0 = blockIdx.y * 64;
    const size_t rowB = (size_t)blockIdx.z * SEQ_T;

    const int tid = threadIdx.x;
    const int w = tid >> 5, lane = tid & 31;
    const int rW = w & 3, cW = w >> 2;
    const int h = lane >> 4, ln16 = lane & 15;

    if (tid < 64) den[tid] = 0.0f;
    __syncthreads();

    v8f num0 = {}, num1 = {};

    for (int sc = 0; sc <= tTile; ++sc) {
        const int s0 = sc * 64;
        v8f a0 = {}, a1 = {};

        // A = q_tile @ k_chunk^T, contraction over D in steps of 32
        for (int dk = 0; dk < D_MODEL / 32; ++dk) {
            {
                int lin = tid * 8;
                int r = lin >> 5, c = lin & 31;
                *(uint4*)&Qs[lin] =
                    *(const uint4*)&qb[(rowB + t0 + r) * D_MODEL + dk * 32 + c];
                *(uint4*)&Ks[lin] =
                    *(const uint4*)&kb[(rowB + s0 + r) * D_MODEL + dk * 32 + c];
            }
            __syncthreads();
            Frag fa, fb0, fb1;
            const int mrow = 16 * rW + ln16;
            fa.q[0] = *(const uint4*)&Qs[mrow * 32 + 8 * h];
            fa.q[1] = *(const uint4*)&Qs[mrow * 32 + 16 + 8 * h];
            const int n0 = 32 * cW + ln16;
            const int n1 = n0 + 16;
            fb0.q[0] = *(const uint4*)&Ks[n0 * 32 + 8 * h];
            fb0.q[1] = *(const uint4*)&Ks[n0 * 32 + 16 + 8 * h];
            fb1.q[0] = *(const uint4*)&Ks[n1 * 32 + 8 * h];
            fb1.q[1] = *(const uint4*)&Ks[n1 * 32 + 16 + 8 * h];
            a0 = wmma_bf16(fa, fb0, a0);
            a1 = wmma_bf16(fa, fb1, a1);
            __syncthreads();
        }

        // causal mask on diagonal chunk, denominator accumulation, A -> LDS bf16
        const bool diag = (sc == tTile);
        #pragma unroll
        for (int t = 0; t < 2; ++t) {
            v8f a = t ? a1 : a0;
            const int scol = 32 * cW + 16 * t + ln16;
            #pragma unroll
            for (int r = 0; r < 8; ++r) {
                const int m = 16 * rW + r + 8 * h;
                float av = a[r];
                if (diag && scol > m) av = 0.0f;
                atomicAdd(&den[m], av);              // ds_add_f32
                Aw[m * 64 + scol] = f2bf(av);
            }
        }
        __syncthreads();

        // num += A @ v_chunk, contraction over s (64) in two steps of 32
        #pragma unroll
        for (int sk = 0; sk < 2; ++sk) {
            #pragma unroll
            for (int i = 0; i < 8; ++i) {   // stage v transposed, coalesced over n
                int lin = tid + 256 * i;
                int kd = lin >> 6, n = lin & 63;
                Vs[n * 32 + kd] =
                    vb[(rowB + s0 + sk * 32 + kd) * D_MODEL + d0 + n];
            }
            __syncthreads();
            Frag fa, fb0, fb1;
            const int mrow = 16 * rW + ln16;
            fa.q[0] = *(const uint4*)&Aw[mrow * 64 + sk * 32 + 8 * h];
            fa.q[1] = *(const uint4*)&Aw[mrow * 64 + sk * 32 + 16 + 8 * h];
            const int n0 = 32 * cW + ln16;
            const int n1 = n0 + 16;
            fb0.q[0] = *(const uint4*)&Vs[n0 * 32 + 8 * h];
            fb0.q[1] = *(const uint4*)&Vs[n0 * 32 + 16 + 8 * h];
            fb1.q[0] = *(const uint4*)&Vs[n1 * 32 + 8 * h];
            fb1.q[1] = *(const uint4*)&Vs[n1 * 32 + 16 + 8 * h];
            num0 = wmma_bf16(fa, fb0, num0);
            num1 = wmma_bf16(fa, fb1, num1);
            __syncthreads();
        }
    }

    // out = num / (den + eps)
    #pragma unroll
    for (int t = 0; t < 2; ++t) {
        v8f nn = t ? num1 : num0;
        const int dc = d0 + 32 * cW + 16 * t + ln16;
        #pragma unroll
        for (int r = 0; r < 8; ++r) {
            const int m = 16 * rW + r + 8 * h;
            const float dd = den[m] + 1e-6f;
            out[(rowB + t0 + m) * D_MODEL + dc] = nn[r] / dd;
        }
    }
}

// ---------------------------------------------------------------------------
extern "C" void kernel_launch(void* const* d_in, const int* in_sizes, int n_in,
                              void* d_out, int out_size, void* d_ws, size_t ws_size,
                              hipStream_t stream) {
    (void)in_sizes; (void)n_in; (void)out_size; (void)ws_size;

    const float* x      = (const float*)d_in[0];
    const float* qkv_w  = (const float*)d_in[1];
    const float* qkv_b  = (const float*)d_in[2];
    const float* gate_w = (const float*)d_in[3];
    const float* gate_b = (const float*)d_in[4];
    const float* ln_g   = (const float*)d_in[5];
    const float* ln_b   = (const float*)d_in[6];
    float* out = (float*)d_out;

    char* ws = (char*)d_ws;
    unsigned short* xnb   = (unsigned short*)ws; ws += (size_t)BT * D_MODEL * 2;
    unsigned short* wqkvb = (unsigned short*)ws; ws += (size_t)D_MODEL * 3 * D_MODEL * 2;
    unsigned short* wgb   = (unsigned short*)ws; ws += (size_t)D_MODEL * D_MODEL * 2;
    unsigned short* qb    = (unsigned short*)ws; ws += (size_t)BT * D_MODEL * 2;
    unsigned short* kb    = (unsigned short*)ws; ws += (size_t)BT * D_MODEL * 2;
    unsigned short* vb    = (unsigned short*)ws;

    const int nqkv = D_MODEL * 3 * D_MODEL;
    cvt_bf16_kernel<<<(nqkv + 255) / 256, 256, 0, stream>>>(qkv_w, wqkvb, nqkv);
    const int ng = D_MODEL * D_MODEL;
    cvt_bf16_kernel<<<(ng + 255) / 256, 256, 0, stream>>>(gate_w, wgb, ng);

    layernorm_bf16_kernel<<<BT, 256, 0, stream>>>(x, ln_g, ln_b, xnb);

    qkvgate_gemm_kernel<<<dim3(BT / 64, D_MODEL / 64, 3), 256, 0, stream>>>(
        xnb, wqkvb, wgb, qkv_b, gate_b, qb, kb, vb);

    linattn_kernel<<<dim3(SEQ_T / 64, D_MODEL / 64, BATCH), 256, 0, stream>>>(
        qb, kb, vb, out);
}
